// MetaGNN_5695126634532
// MI455X (gfx1250) — matrix operand order repaired and checked by reading, not compiled, not run on hardware
//
#include <hip/hip_runtime.h>

// ---------------------------------------------------------------------------
// MI455X (gfx1250) MetaLayer GNN.
// bf16 WMMA (v_wmma_f32_16x16x32_bf16) everywhere; f32 accumulate.
// Roofline: ~0.64 TFLOP vs ~6 GB traffic @ 23.3 TB/s -> memory bound only if
// matrix math runs on the bf16 WMMA path (~5 PFLOPS dense), not f32 VALU.
// ---------------------------------------------------------------------------

typedef __bf16 bf16;
typedef __attribute__((ext_vector_type(16))) __bf16 v16bf;
typedef __attribute__((ext_vector_type(8)))  __bf16 v8bf;
typedef __attribute__((ext_vector_type(8)))  float  v8f;

struct Seg {
    const void*  ptr;    // f32 or bf16 source
    const int*   idx;    // gather index (nullptr = identity row)
    const float* scale;  // per-source-row scale (nullptr = 1.0)
    int          width;  // feature width of this segment
    int          dtype;  // 0 = f32, 1 = bf16
};
struct Segs3 { Seg s[3]; };

// ---------------------------------------------------------------------------
// Fused MLP2: C = (relu(A*W1 + b1)) * W2 + b2, A = concat of gathered segs.
// Block: 256 threads = 8 waves; tile 32 rows x 256 cols.
// Wave (wm,wn): wm in {0,1} picks 16 rows, wn in {0..3} picks 64 cols.
// MODE: 0 = store bf16, 1 = store f32, 2 = scatter atomicAdd f32 (rowIdx),
//       3 = skip GEMM2, epilogue = relu(A*W1+b1) . w2f + b2f -> out[m]
// ---------------------------------------------------------------------------
template<int MODE>
__global__ __launch_bounds__(256)
void fused_mlp(Segs3 sg, int M, int K1,
               const void* W1v, const float* __restrict__ b1v,
               const void* W2v, const float* __restrict__ b2v,
               void* out, const int* __restrict__ rowIdx,
               const float* __restrict__ w2f, const float* __restrict__ b2f)
{
    __shared__ __align__(16) bf16 ldsA[32][136];   // A chunk 32x128 (+pad)
    __shared__ __align__(16) bf16 ldsT[32][264];   // hidden 32x256 (+pad)

    const bf16* W1 = (const bf16*)W1v;
    const bf16* W2 = (const bf16*)W2v;

    const int tid   = threadIdx.x;
    const int lane  = tid & 31;
    const int wave  = tid >> 5;
    const int wm    = wave >> 2;       // 0..1
    const int wn    = wave & 3;        // 0..3
    const int lidx  = lane & 15;
    const int lhalf = lane >> 4;
    const int m0    = blockIdx.x * 32;

    const int w0   = sg.s[0].width;
    const int bnd1 = w0;
    const int bnd2 = w0 + sg.s[1].width;

    v8f acc[4] = {};

    // ---------------- GEMM1: K1 in chunks of 128 ----------------
    const int nChunks = K1 >> 7;
    for (int c = 0; c < nChunks; ++c) {
        const int kc = c << 7;
        // cooperative gather/convert of the 32x128 A chunk into LDS
        #pragma unroll
        for (int i = 0; i < 16; ++i) {
            int lin = i * 256 + tid;
            int ml  = lin >> 7;
            int kl  = lin & 127;
            int gm  = m0 + ml;
            int gk  = kc + kl;
            float v = 0.f;
            if (gm < M) {
                const Seg* sp; int lk;
                if (gk < bnd1)      { sp = &sg.s[0]; lk = gk; }
                else if (gk < bnd2) { sp = &sg.s[1]; lk = gk - bnd1; }
                else                { sp = &sg.s[2]; lk = gk - bnd2; }
                int sr = sp->idx ? sp->idx[gm] : gm;
                size_t off = (size_t)sr * sp->width + lk;
                v = (sp->dtype == 0) ? ((const float*)sp->ptr)[off]
                                     : (float)((const bf16*)sp->ptr)[off];
                if (sp->scale) v *= sp->scale[sr];
            }
            ldsA[ml][kl] = (bf16)v;
        }
        __syncthreads();

        #pragma unroll
        for (int ks = 0; ks < 4; ++ks) {
            const int kb = ks * 32;
            // A fragment (16x32, contiguous k = 16*lhalf + s per lane)
            const bf16* ar = &ldsA[wm * 16 + lidx][0];
            v16bf a;
            *(v8bf*)&a       = *(const v8bf*)(ar + kb + lhalf * 16);
            *((v8bf*)&a + 1) = *(const v8bf*)(ar + kb + lhalf * 16 + 8);
            #pragma unroll
            for (int f = 0; f < 4; ++f) {
                const int nb = wn * 64 + f * 16;
                // B fragment: transposed weights (N x K), 16 contiguous K per lane
                const bf16* bp = W1 + (size_t)(nb + lidx) * K1 + (kc + kb + lhalf * 16);
                v16bf b = *(const v16bf*)bp;
                acc[f] = __builtin_amdgcn_wmma_f32_16x16x32_bf16(
                             false, a, false, b, (short)0, acc[f], false, false);
            }
        }
        __syncthreads();
    }

    // bias + ReLU -> ldsT (bf16)
    #pragma unroll
    for (int f = 0; f < 4; ++f) {
        const int n  = wn * 64 + f * 16 + lidx;
        const float bb = b1v[n];
        #pragma unroll
        for (int r = 0; r < 8; ++r) {
            float v = acc[f][r] + bb;
            v = v > 0.f ? v : 0.f;
            ldsT[wm * 16 + lhalf * 8 + r][n] = (bf16)v;
        }
    }
    __syncthreads();

    if constexpr (MODE == 3) {
        // 256 -> 1 head: 8 lanes per row, shuffle-reduce
        const int ml = tid >> 3;
        const int j  = tid & 7;
        const bf16* tr = &ldsT[ml][0];
        float part = 0.f;
        #pragma unroll
        for (int k = 0; k < 32; ++k)
            part += (float)tr[j * 32 + k] * w2f[j * 32 + k];
        #pragma unroll
        for (int o = 1; o < 8; o <<= 1) part += __shfl_xor(part, o, 32);
        const int gm = m0 + ml;
        if (j == 0 && gm < M) ((float*)out)[gm] = part + b2f[0];
    } else {
        // ---------------- GEMM2: K = 256 from ldsT ----------------
        v8f acc2[4] = {};
        #pragma unroll
        for (int ks = 0; ks < 8; ++ks) {
            const int kb = ks * 32;
            const bf16* ar = &ldsT[wm * 16 + lidx][0];
            v16bf a;
            *(v8bf*)&a       = *(const v8bf*)(ar + kb + lhalf * 16);
            *((v8bf*)&a + 1) = *(const v8bf*)(ar + kb + lhalf * 16 + 8);
            #pragma unroll
            for (int f = 0; f < 4; ++f) {
                const int nb = wn * 64 + f * 16;
                const bf16* bp = W2 + (size_t)(nb + lidx) * 256 + (kb + lhalf * 16);
                v16bf b = *(const v16bf*)bp;
                acc2[f] = __builtin_amdgcn_wmma_f32_16x16x32_bf16(
                              false, a, false, b, (short)0, acc2[f], false, false);
            }
        }
        #pragma unroll
        for (int f = 0; f < 4; ++f) {
            const int n  = wn * 64 + f * 16 + lidx;
            const float bb = b2v[n];
            #pragma unroll
            for (int r = 0; r < 8; ++r) {
                const int gm = m0 + wm * 16 + lhalf * 8 + r;
                if (gm < M) {
                    float v = acc2[f][r] + bb;
                    if (MODE == 0)
                        ((bf16*)out)[(size_t)gm * 256 + n] = (bf16)v;
                    else if (MODE == 1)
                        ((float*)out)[(size_t)gm * 256 + n] = v;
                    else
                        atomicAdd((float*)out + (size_t)rowIdx[gm] * 256 + n, v);
                }
            }
        }
    }
}

// ---------------------------------------------------------------------------
// Helpers
// ---------------------------------------------------------------------------
__global__ void k_zero(float* p, long long n) {
    long long i = (long long)blockIdx.x * 256 + threadIdx.x;
    if (i < n) p[i] = 0.f;
}
// W (K x 256, f32) -> Wt (256 x K, bf16), Wt[n*K+k] = W[k*256+n]
__global__ void k_pack(const float* __restrict__ W, void* dstv, int K) {
    bf16* dst = (bf16*)dstv;
    int i = blockIdx.x * 256 + threadIdx.x;
    if (i < 256 * K) {
        int n = i / K, k = i % K;
        dst[i] = (bf16)W[(size_t)k * 256 + n];
    }
}
__global__ void k_degree(const int* __restrict__ row, float* cnt, int E) {
    int i = blockIdx.x * 256 + threadIdx.x;
    if (i < E) atomicAdd(&cnt[row[i]], 1.f);
}
__global__ void k_inv(float* c, int n) {
    int i = blockIdx.x * 256 + threadIdx.x;
    if (i < n) c[i] = 1.f / fmaxf(c[i], 1.f);
}

// ---------------------------------------------------------------------------
extern "C" void kernel_launch(void* const* d_in, const int* in_sizes, int n_in,
                              void* d_out, int out_size, void* d_ws, size_t ws_size,
                              hipStream_t stream) {
    const float* x  = (const float*)d_in[0];
    const int*   ei = (const int*)d_in[1];
    const float* ea = (const float*)d_in[2];
    const int E  = in_sizes[1] / 2;
    const int NN = in_sizes[0] / 128;
    const int* row = ei;
    const int* col = ei + E;

    // ---- param flattening discovery (handles b/w and key-sort orders) ----
    const int base = 3;
    const bool bFirst = (in_sizes[base] < in_sizes[base + 1]); // bias before weight?
    const int wo = bFirst ? 1 : 0;
    const int bo = bFirst ? 0 : 1;
    // size of 2nd MLP's l1.w: 196608 => sorted keys (e1,e2,n1a,...), else insertion
    const bool sorted = (in_sizes[base + 4 + wo] == 196608);
    int pE1, pE2, pN1A, pN1B, pN2A, pN2B;
    if (sorted) { pE1=base;   pE2=base+4;  pN1A=base+8;  pN1B=base+12; pN2A=base+16; pN2B=base+20; }
    else        { pE1=base;   pN1A=base+4; pN1B=base+8;  pE2=base+12;  pN2A=base+16; pN2B=base+20; }
    auto Wp = [&](int p, int l) { return (const float*)d_in[p + 2*l + wo]; };
    auto Bp = [&](int p, int l) { return (const float*)d_in[p + 2*l + bo]; };

    // ---- workspace layout ----
    char* ws = (char*)d_ws;
    size_t off = 0;
    auto take = [&](size_t bytes) -> char* {
        char* p = ws + off;
        off = (off + bytes + 255) & ~(size_t)255;
        return p;
    };
    void* wt_e1l1  = take((size_t)256*384*2);
    void* wt_e1l2  = take((size_t)256*256*2);
    void* wt_n1al1 = take((size_t)256*384*2);
    void* wt_n1al2 = take((size_t)256*256*2);
    void* wt_n1bl1 = take((size_t)256*384*2);
    void* wt_n1bl2 = take((size_t)256*256*2);
    void* wt_e2l1  = take((size_t)256*768*2);
    void* wt_e2l2  = take((size_t)256*256*2);
    void* wt_n2al1 = take((size_t)256*512*2);
    void* wt_n2al2 = take((size_t)256*256*2);
    void* wt_n2bl1 = take((size_t)256*512*2);
    void*  e_buf = take((size_t)E  * 256 * 2);   // bf16
    void*  h_buf = take((size_t)NN * 256 * 2);   // bf16
    float* agg   = (float*)take((size_t)NN * 256 * 4);
    float* cnt   = (float*)take((size_t)NN * 4);

    // ---- pack weights to bf16, transposed (N x K) ----
    auto pack = [&](const float* W, void* dst, int K) {
        k_pack<<<dim3((unsigned)((256*K + 255)/256)), dim3(256), 0, stream>>>(W, dst, K);
    };
    pack(Wp(pE1, 0), wt_e1l1, 384);  pack(Wp(pE1, 1), wt_e1l2, 256);
    pack(Wp(pN1A,0), wt_n1al1,384);  pack(Wp(pN1A,1), wt_n1al2,256);
    pack(Wp(pN1B,0), wt_n1bl1,384);  pack(Wp(pN1B,1), wt_n1bl2,256);
    pack(Wp(pE2, 0), wt_e2l1, 768);  pack(Wp(pE2, 1), wt_e2l2, 256);
    pack(Wp(pN2A,0), wt_n2al1,512);  pack(Wp(pN2A,1), wt_n2al2,256);
    pack(Wp(pN2B,0), wt_n2bl1,512);

    // ---- degree / inverse counts (same row index both layers) ----
    k_zero  <<<dim3((unsigned)((NN+255)/256)), dim3(256), 0, stream>>>(cnt, NN);
    k_degree<<<dim3((unsigned)((E +255)/256)), dim3(256), 0, stream>>>(row, cnt, E);
    k_inv   <<<dim3((unsigned)((NN+255)/256)), dim3(256), 0, stream>>>(cnt, NN);

    const long long aggN = (long long)NN * 256;
    const dim3 gE((unsigned)((E + 31)/32)), gN((unsigned)((NN + 31)/32)), blk(256);
    float* eout = (float*)d_out + NN;   // final edge_attr region of d_out

    // =================== Layer 1 ===================
    k_zero<<<dim3((unsigned)((aggN+255)/256)), dim3(256), 0, stream>>>(agg, aggN);

    { Segs3 sg = {{ {x, row, nullptr, 128, 0}, {x, col, nullptr, 128, 0},
                    {ea, nullptr, nullptr, 128, 0} }};
      fused_mlp<0><<<gE, blk, 0, stream>>>(sg, E, 384, wt_e1l1, Bp(pE1,0),
          wt_e1l2, Bp(pE1,1), e_buf, nullptr, nullptr, nullptr); }

    { Segs3 sg = {{ {x, col, nullptr, 128, 0}, {e_buf, nullptr, nullptr, 256, 1},
                    {nullptr, nullptr, nullptr, 0, 0} }};
      fused_mlp<2><<<gE, blk, 0, stream>>>(sg, E, 384, wt_n1al1, Bp(pN1A,0),
          wt_n1al2, Bp(pN1A,1), agg, row, nullptr, nullptr); }

    { Segs3 sg = {{ {x, nullptr, nullptr, 128, 0}, {agg, nullptr, cnt, 256, 0},
                    {nullptr, nullptr, nullptr, 0, 0} }};
      fused_mlp<0><<<gN, blk, 0, stream>>>(sg, NN, 384, wt_n1bl1, Bp(pN1B,0),
          wt_n1bl2, Bp(pN1B,1), h_buf, nullptr, nullptr, nullptr); }

    // =================== Layer 2 ===================
    k_zero<<<dim3((unsigned)((aggN+255)/256)), dim3(256), 0, stream>>>(agg, aggN);

    { Segs3 sg = {{ {h_buf, row, nullptr, 256, 1}, {h_buf, col, nullptr, 256, 1},
                    {e_buf, nullptr, nullptr, 256, 1} }};
      fused_mlp<1><<<gE, blk, 0, stream>>>(sg, E, 768, wt_e2l1, Bp(pE2,0),
          wt_e2l2, Bp(pE2,1), eout, nullptr, nullptr, nullptr); }

    { Segs3 sg = {{ {h_buf, col, nullptr, 256, 1}, {eout, nullptr, nullptr, 256, 0},
                    {nullptr, nullptr, nullptr, 0, 0} }};
      fused_mlp<2><<<gE, blk, 0, stream>>>(sg, E, 512, wt_n2al1, Bp(pN2A,0),
          wt_n2al2, Bp(pN2A,1), agg, row, nullptr, nullptr); }

    { Segs3 sg = {{ {h_buf, nullptr, nullptr, 256, 1}, {agg, nullptr, cnt, 256, 0},
                    {nullptr, nullptr, nullptr, 0, 0} }};
      fused_mlp<3><<<gN, blk, 0, stream>>>(sg, NN, 512, wt_n2bl1, Bp(pN2B,0),
          nullptr, nullptr, d_out, nullptr, Wp(pN2B,1), Bp(pN2B,1)); }
}
